// GraphNetBlock_88098369175666
// MI455X (gfx1250) — compile-verified, matrix operand order checked by reading, not run on hardware
//
#include <hip/hip_runtime.h>

// ---------------------------------------------------------------------------
// GraphNetBlock for MI455X (gfx1250, wave32, WMMA).
// Heavy GEMMs run as v_wmma_f32_16x16x32_bf16 with f32 accumulation.
// ---------------------------------------------------------------------------

#define DLAT 128      // latent dim
#define NT   8        // 128 / 16 output-column tiles

typedef __attribute__((ext_vector_type(16))) __bf16       v16bf;
typedef __attribute__((ext_vector_type(8)))  float        v8f;
typedef __attribute__((ext_vector_type(4)))  unsigned int v4u;
typedef __attribute__((ext_vector_type(4)))  float        v4f;

union FragAB { v16bf v; v4u q[2]; };           // 16 bf16 per lane (A or B of 16x16x32)
union FragC  { v8f   v; float f[8]; };         // 8 f32 per lane (C/D 16x16)

__device__ __forceinline__ unsigned short f2bf(float f) {
    unsigned int u = __float_as_uint(f);
    u += 0x7FFFu + ((u >> 16) & 1u);           // round-to-nearest-even
    return (unsigned short)(u >> 16);
}

// --------------------------- prep kernels ----------------------------------

__global__ void __launch_bounds__(256) zero_kernel(v4f* __restrict__ p, int n4) {
    int i = blockIdx.x * 256 + threadIdx.x;
    if (i < n4) { v4f z = {0.f, 0.f, 0.f, 0.f}; p[i] = z; }
}

// Repack fp32 weight W[K][128] into bf16 B-fragments:
// packed[(((ks*NT + nt)*32 + lane)*16 + e]  with
//   k = ks*32 + (e>=8 ? 16 : 0) + (lane>>4)*8 + (e&7),  n = nt*16 + (lane&15)
__global__ void __launch_bounds__(256) pack_weights(
    const float* __restrict__ w, unsigned short* __restrict__ out, int K) {
    int p = blockIdx.x * 256 + threadIdx.x;
    if (p >= K * DLAT) return;
    int e    =  p        & 15;
    int lane = (p >> 4)  & 31;
    int nt   = (p >> 9)  & 7;
    int ks   =  p >> 12;
    int k = ks * 32 + ((e >> 3) << 4) + ((lane >> 4) << 3) + (e & 7);
    int n = nt * 16 + (lane & 15);
    out[p] = f2bf(w[(size_t)k * DLAT + n]);
}

// --------------------------- edge MLP + scatter ----------------------------
// block = 256 threads = 8 waves; each block handles 128 edges; each wave one
// 16-row strip. Layer1: [16x384]@[384x128], Layer2: [16x128]@[128x128].

__global__ void __launch_bounds__(256) edge_mlp_kernel(
    const float* __restrict__ node_latent,
    const float* __restrict__ edge_latent,
    const int*   __restrict__ senders,
    const int*   __restrict__ receivers,
    const unsigned short* __restrict__ w1p,   // [12][8][32][16] bf16
    const float* __restrict__ b1,
    const unsigned short* __restrict__ w2p,   // [4][8][32][16] bf16
    const float* __restrict__ b2,
    float* __restrict__ new_edge,             // E x 128 out
    float* __restrict__ agg,                  // N x 128 accum (zeroed)
    int nEdges)
{
    extern __shared__ unsigned short smem[];
    unsigned short* sIn = smem;                // 128 x 384 bf16 (96 KB)
    unsigned short* sH  = smem + 128 * 384;    // 128 x 128 bf16 (32 KB)
    int* sS = (int*)(smem + 128 * 384 + 128 * 128);  // 128 sender idx
    int* sR = sS + 128;                              // 128 receiver idx

    const int tid   = threadIdx.x;
    const int eBase = blockIdx.x * 128;

    // Stage the block's edge endpoints once (kills 32x-redundant index gathers)
    if (tid < 128) {
        int e = eBase + tid; if (e >= nEdges) e = nEdges - 1;
        sS[tid] = senders[e];
    } else {
        int e = eBase + (tid - 128); if (e >= nEdges) e = nEdges - 1;
        sR[tid - 128] = receivers[e];
    }
    __syncthreads();

    // Stage gathered input as bf16: cols [0,128)=edge, [128,256)=sender, [256,384)=receiver
    for (int wi = tid; wi < 128 * 96; wi += 256) {
        int row = wi / 96;
        int col = (wi % 96) * 4;
        int e = eBase + row; if (e >= nEdges) e = nEdges - 1;
        const float* src;
        if (col < 128)      src = edge_latent + (size_t)e * DLAT + col;
        else if (col < 256) src = node_latent + (size_t)sS[row] * DLAT + (col - 128);
        else                src = node_latent + (size_t)sR[row] * DLAT + (col - 256);
        v4f x = *(const v4f*)src;
        unsigned int lo = (unsigned int)f2bf(x.x) | ((unsigned int)f2bf(x.y) << 16);
        unsigned int hi = (unsigned int)f2bf(x.z) | ((unsigned int)f2bf(x.w) << 16);
        unsigned int* d = (unsigned int*)(sIn + row * 384 + col);
        d[0] = lo; d[1] = hi;
    }
    __syncthreads();

    const int lane  = tid & 31;
    const int wave  = tid >> 5;
    const int laneM = lane & 15;
    const int laneH = lane >> 4;
    const int m0    = wave * 16;

    // ---- layer 1: 12 k-steps of 16x16x32 bf16 WMMA across 8 n-tiles ----
    FragC acc[NT];
    #pragma unroll
    for (int nt = 0; nt < NT; ++nt)
        #pragma unroll
        for (int r = 0; r < 8; ++r) acc[nt].f[r] = 0.f;

    const v4u* aRow = (const v4u*)(sIn + (m0 + laneM) * 384);   // 48 chunks/row
    for (int ks = 0; ks < 12; ++ks) {
        FragAB a;
        a.q[0] = aRow[ks * 4 + laneH];        // K = ks*32 + laneH*8 + 0..7
        a.q[1] = aRow[ks * 4 + laneH + 2];    // K = +16
        const v4u* wp = (const v4u*)w1p + ((size_t)(ks * NT) * 32 + lane) * 2;
        #pragma unroll
        for (int nt = 0; nt < NT; ++nt) {
            FragAB b;
            b.q[0] = wp[nt * 64];
            b.q[1] = wp[nt * 64 + 1];
            acc[nt].v = __builtin_amdgcn_wmma_f32_16x16x32_bf16(
                false, a.v, false, b.v, (short)0, acc[nt].v, false, false);
        }
    }

    // bias + ReLU -> bf16 hidden in LDS (each wave writes only its own strip)
    #pragma unroll
    for (int nt = 0; nt < NT; ++nt) {
        float bias = b1[nt * 16 + laneM];
        #pragma unroll
        for (int r = 0; r < 8; ++r) {
            float v = acc[nt].f[r] + bias;
            v = v > 0.f ? v : 0.f;
            int m = m0 + laneH * 8 + r;
            sH[m * DLAT + nt * 16 + laneM] = f2bf(v);
        }
    }
    // wave-private LDS strip: same-wave RAW handled by s_wait_dscnt, no barrier

    // Prefetch the fp32 residual rows (global_prefetch_b8) so the epilogue's
    // re-read of edge_latent overlaps the layer-2 WMMA loop.
    #pragma unroll
    for (int rr = 0; rr < 16; ++rr) {
        int e = eBase + m0 + rr;
        if (e < nEdges)
            __builtin_prefetch(edge_latent + (size_t)e * DLAT + lane * 4, 0, 0);
    }

    // ---- layer 2: 4 k-steps ----
    FragC acc2[NT];
    #pragma unroll
    for (int nt = 0; nt < NT; ++nt)
        #pragma unroll
        for (int r = 0; r < 8; ++r) acc2[nt].f[r] = 0.f;

    const v4u* hRow = (const v4u*)(sH + (m0 + laneM) * DLAT);   // 16 chunks/row
    for (int ks = 0; ks < 4; ++ks) {
        FragAB a;
        a.q[0] = hRow[ks * 4 + laneH];
        a.q[1] = hRow[ks * 4 + laneH + 2];
        const v4u* wp = (const v4u*)w2p + ((size_t)(ks * NT) * 32 + lane) * 2;
        #pragma unroll
        for (int nt = 0; nt < NT; ++nt) {
            FragAB b;
            b.q[0] = wp[nt * 64];
            b.q[1] = wp[nt * 64 + 1];
            acc2[nt].v = __builtin_amdgcn_wmma_f32_16x16x32_bf16(
                false, a.v, false, b.v, (short)0, acc2[nt].v, false, false);
        }
    }

    float bias2[NT];
    #pragma unroll
    for (int nt = 0; nt < NT; ++nt) bias2[nt] = b2[nt * 16 + laneM];

    // epilogue: residual, store new_edge, scatter-add into agg (L2 fp32 atomics)
    #pragma unroll
    for (int r = 0; r < 8; ++r) {
        int m = m0 + laneH * 8 + r;
        int e = eBase + m;
        if (e < nEdges) {
            const float* el = edge_latent + (size_t)e * DLAT;
            float*       op = new_edge    + (size_t)e * DLAT;
            float*       ap = agg + (size_t)sR[m] * DLAT;
            #pragma unroll
            for (int nt = 0; nt < NT; ++nt) {
                int col = nt * 16 + laneM;
                float v = acc2[nt].f[r] + bias2[nt] + el[col];
                op[col] = v;
                unsafeAtomicAdd(ap + col, v);
            }
        }
    }
}

// --------------------------- node MLP --------------------------------------

__global__ void __launch_bounds__(256) node_mlp_kernel(
    const float* __restrict__ node_latent,
    const float* __restrict__ agg,
    const unsigned short* __restrict__ w1p,   // [8][8][32][16] bf16
    const float* __restrict__ b1,
    const unsigned short* __restrict__ w2p,   // [4][8][32][16] bf16
    const float* __restrict__ b2,
    float* __restrict__ new_node,
    int nNodes)
{
    extern __shared__ unsigned short smem[];
    unsigned short* sIn = smem;                // 128 x 256 bf16 (64 KB)
    unsigned short* sH  = smem + 128 * 256;    // 128 x 128 bf16 (32 KB)

    const int tid   = threadIdx.x;
    const int nBase = blockIdx.x * 128;

    for (int wi = tid; wi < 128 * 64; wi += 256) {
        int row = wi / 64;
        int col = (wi % 64) * 4;
        int n = nBase + row; if (n >= nNodes) n = nNodes - 1;
        const float* src = (col < 128)
            ? node_latent + (size_t)n * DLAT + col
            : agg         + (size_t)n * DLAT + (col - 128);
        v4f x = *(const v4f*)src;
        unsigned int lo = (unsigned int)f2bf(x.x) | ((unsigned int)f2bf(x.y) << 16);
        unsigned int hi = (unsigned int)f2bf(x.z) | ((unsigned int)f2bf(x.w) << 16);
        unsigned int* d = (unsigned int*)(sIn + row * 256 + col);
        d[0] = lo; d[1] = hi;
    }
    __syncthreads();

    const int lane  = tid & 31;
    const int wave  = tid >> 5;
    const int laneM = lane & 15;
    const int laneH = lane >> 4;
    const int m0    = wave * 16;

    FragC acc[NT];
    #pragma unroll
    for (int nt = 0; nt < NT; ++nt)
        #pragma unroll
        for (int r = 0; r < 8; ++r) acc[nt].f[r] = 0.f;

    const v4u* aRow = (const v4u*)(sIn + (m0 + laneM) * 256);   // 32 chunks/row
    for (int ks = 0; ks < 8; ++ks) {
        FragAB a;
        a.q[0] = aRow[ks * 4 + laneH];
        a.q[1] = aRow[ks * 4 + laneH + 2];
        const v4u* wp = (const v4u*)w1p + ((size_t)(ks * NT) * 32 + lane) * 2;
        #pragma unroll
        for (int nt = 0; nt < NT; ++nt) {
            FragAB b;
            b.q[0] = wp[nt * 64];
            b.q[1] = wp[nt * 64 + 1];
            acc[nt].v = __builtin_amdgcn_wmma_f32_16x16x32_bf16(
                false, a.v, false, b.v, (short)0, acc[nt].v, false, false);
        }
    }

    #pragma unroll
    for (int nt = 0; nt < NT; ++nt) {
        float bias = b1[nt * 16 + laneM];
        #pragma unroll
        for (int r = 0; r < 8; ++r) {
            float v = acc[nt].f[r] + bias;
            v = v > 0.f ? v : 0.f;
            int m = m0 + laneH * 8 + r;
            sH[m * DLAT + nt * 16 + laneM] = f2bf(v);
        }
    }

    // Prefetch fp32 residual rows to overlap layer-2 compute
    #pragma unroll
    for (int rr = 0; rr < 16; ++rr) {
        int n = nBase + m0 + rr;
        if (n < nNodes)
            __builtin_prefetch(node_latent + (size_t)n * DLAT + lane * 4, 0, 0);
    }

    FragC acc2[NT];
    #pragma unroll
    for (int nt = 0; nt < NT; ++nt)
        #pragma unroll
        for (int r = 0; r < 8; ++r) acc2[nt].f[r] = 0.f;

    const v4u* hRow = (const v4u*)(sH + (m0 + laneM) * DLAT);
    for (int ks = 0; ks < 4; ++ks) {
        FragAB a;
        a.q[0] = hRow[ks * 4 + laneH];
        a.q[1] = hRow[ks * 4 + laneH + 2];
        const v4u* wp = (const v4u*)w2p + ((size_t)(ks * NT) * 32 + lane) * 2;
        #pragma unroll
        for (int nt = 0; nt < NT; ++nt) {
            FragAB b;
            b.q[0] = wp[nt * 64];
            b.q[1] = wp[nt * 64 + 1];
            acc2[nt].v = __builtin_amdgcn_wmma_f32_16x16x32_bf16(
                false, a.v, false, b.v, (short)0, acc2[nt].v, false, false);
        }
    }

    float bias2[NT];
    #pragma unroll
    for (int nt = 0; nt < NT; ++nt) bias2[nt] = b2[nt * 16 + laneM];

    #pragma unroll
    for (int r = 0; r < 8; ++r) {
        int n = nBase + m0 + laneH * 8 + r;
        if (n < nNodes) {
            const float* nl = node_latent + (size_t)n * DLAT;
            float*       op = new_node    + (size_t)n * DLAT;
            #pragma unroll
            for (int nt = 0; nt < NT; ++nt) {
                int col = nt * 16 + laneM;
                op[col] = acc2[nt].f[r] + bias2[nt] + nl[col];
            }
        }
    }
}

// --------------------------- host launch -----------------------------------

extern "C" void kernel_launch(void* const* d_in, const int* in_sizes, int n_in,
                              void* d_out, int out_size, void* d_ws, size_t ws_size,
                              hipStream_t stream)
{
    const float* node_latent = (const float*)d_in[0];
    const float* edge_latent = (const float*)d_in[1];
    const int*   senders     = (const int*)  d_in[2];
    const int*   receivers   = (const int*)  d_in[3];
    const float* ew1 = (const float*)d_in[4];
    const float* eb1 = (const float*)d_in[5];
    const float* ew2 = (const float*)d_in[6];
    const float* eb2 = (const float*)d_in[7];
    const float* nw1 = (const float*)d_in[8];
    const float* nb1 = (const float*)d_in[9];
    const float* nw2 = (const float*)d_in[10];
    const float* nb2 = (const float*)d_in[11];

    const int N = in_sizes[0] / DLAT;
    const int E = in_sizes[1] / DLAT;

    float* new_node = (float*)d_out;                       // N x 128
    float* new_edge = (float*)d_out + (size_t)N * DLAT;    // E x 128

    // workspace: [agg f32 NxD][w1e][w2e][w1n][w2n] (bf16 packed fragments)
    char* ws = (char*)d_ws;
    float* agg = (float*)ws;
    size_t aggBytes = (size_t)N * DLAT * sizeof(float);    // 256-aligned for N=100000
    unsigned short* w1e = (unsigned short*)(ws + aggBytes);
    unsigned short* w2e = w1e + 3 * DLAT * DLAT;
    unsigned short* w1n = w2e + DLAT * DLAT;
    unsigned short* w2n = w1n + 2 * DLAT * DLAT;
    (void)ws_size; (void)n_in; (void)out_size;

    int n4 = (N * DLAT) / 4;
    zero_kernel<<<(n4 + 255) / 256, 256, 0, stream>>>((v4f*)agg, n4);

    pack_weights<<<(3 * DLAT * DLAT + 255) / 256, 256, 0, stream>>>(ew1, w1e, 3 * DLAT);
    pack_weights<<<(DLAT * DLAT + 255) / 256, 256, 0, stream>>>(ew2, w2e, DLAT);
    pack_weights<<<(2 * DLAT * DLAT + 255) / 256, 256, 0, stream>>>(nw1, w1n, 2 * DLAT);
    pack_weights<<<(DLAT * DLAT + 255) / 256, 256, 0, stream>>>(nw2, w2n, DLAT);

    size_t smemE = (size_t)(128 * 384 + 128 * 128) * sizeof(unsigned short)
                 + 256 * sizeof(int);                                        // ~129 KB
    edge_mlp_kernel<<<(E + 127) / 128, 256, smemE, stream>>>(
        node_latent, edge_latent, senders, receivers,
        w1e, eb1, w2e, eb2, new_edge, agg, E);

    size_t smemN = (size_t)(128 * 256 + 128 * 128) * sizeof(unsigned short); // 96 KB
    node_mlp_kernel<<<(N + 127) / 128, 256, smemN, stream>>>(
        node_latent, agg, w1n, nb1, w2n, nb2, new_node, N);
}
